// EdgeAwareSelfAttention_60138132078619
// MI455X (gfx1250) — matrix operand order repaired and checked by reading, not compiled, not run on hardware
//
#include <hip/hip_runtime.h>
#include <hip/hip_bf16.h>

// ---------------------------------------------------------------------------
// EdgeAwareSelfAttention on gfx1250 (MI455X). All matmuls via
// v_wmma_f32_16x16x32_bf16. Bandwidth-bound on the one-shot 302 MB rel_emb
// stream: ebias_kernel uses coalesced non-temporal loads + LDS staging.
// ---------------------------------------------------------------------------

typedef __attribute__((ext_vector_type(16))) __bf16 v16bf;
typedef __attribute__((ext_vector_type(8)))  __bf16 v8bf;
typedef __attribute__((ext_vector_type(4)))  __bf16 v4bf;
typedef __attribute__((ext_vector_type(8)))  float  v8f;
typedef __attribute__((ext_vector_type(4)))  float  f32x4;

static constexpr int Bc = 2;
static constexpr int Nc = 384;
static constexpr int Dc = 256;
static constexpr int Hc = 8;
static constexpr int HDc = 32;
static constexpr int WPB = 4;          // waves per block (blockDim = 128)

__device__ __forceinline__ v8f wmma_bf16(v16bf a, v16bf b, v8f c) {
  return __builtin_amdgcn_wmma_f32_16x16x32_bf16(
      /*neg_a=*/false, a, /*neg_b=*/false, b,
      /*c_mod=*/(short)0, c, /*reuse_a=*/false, /*reuse_b=*/false);
}

// A fragment: 16x32 (MxK) bf16, from row-major fp32, base = &A[row0][k0].
// Layout: lane m=lane&15; lanes0-15 K 0..15, lanes16-31 K 16..31 (contiguous 16 floats).
__device__ __forceinline__ v16bf load_a_frag(const float* base, int ld, int lane) {
  int m  = lane & 15;
  int kb = (lane & 16) ? 16 : 0;
  const float* r = base + (long)m * ld + kb;
  f32x4 l0 = *(const f32x4*)(r + 0);
  f32x4 l1 = *(const f32x4*)(r + 4);
  f32x4 h0 = *(const f32x4*)(r + 8);
  f32x4 h1 = *(const f32x4*)(r + 12);
  v16bf a;
#pragma unroll
  for (int j = 0; j < 4; ++j) {
    a[j]      = (__bf16)l0[j];
    a[4 + j]  = (__bf16)l1[j];
    a[8 + j]  = (__bf16)h0[j];
    a[12 + j] = (__bf16)h1[j];
  }
  return a;
}

// B fragment: 32x16 (KxN) where B[k][n] = W[n][k] (W row-major, base=&W[n0][k0]).
// Lane col n=lane&15; lanes0-15 K 0..15, lanes16-31 K 16..31 (contiguous 16 floats).
__device__ __forceinline__ v16bf load_bT_frag(const float* base, int ld, int lane) {
  int n  = lane & 15;
  int ko = (lane & 16) ? 16 : 0;
  const float* r = base + (long)n * ld + ko;
  f32x4 q0 = *(const f32x4*)(r + 0);
  f32x4 q1 = *(const f32x4*)(r + 4);
  f32x4 q2 = *(const f32x4*)(r + 8);
  f32x4 q3 = *(const f32x4*)(r + 12);
  v16bf b;
#pragma unroll
  for (int j = 0; j < 4; ++j) {
    b[j]      = (__bf16)q0[j];
    b[4 + j]  = (__bf16)q1[j];
    b[8 + j]  = (__bf16)q2[j];
    b[12 + j] = (__bf16)q3[j];
  }
  return b;
}

// B fragment from true row-major B[k][n], stride ld (strided gather).
__device__ __forceinline__ v16bf load_b_rm_frag(const float* base, int ld, int lane) {
  int n  = lane & 15;
  int ko = (lane & 16) ? 16 : 0;
  v16bf b;
#pragma unroll
  for (int e = 0; e < 16; ++e)
    b[e] = (__bf16)base[(long)(ko + e) * ld + n];
  return b;
}

__device__ __forceinline__ int wave_id() {
  return blockIdx.x * (blockDim.x >> 5) + (threadIdx.x >> 5);
}

// ---------------- Kernel 1: Q,K,V projections (y = h @ W.T + b) -------------
__global__ void qkv_kernel(const float* __restrict__ h,
                           const float* __restrict__ Wq, const float* __restrict__ bq,
                           const float* __restrict__ Wk, const float* __restrict__ bk,
                           const float* __restrict__ Wv, const float* __restrict__ bv,
                           float* __restrict__ q, float* __restrict__ k,
                           float* __restrict__ v) {
  const int MT = (Bc * Nc) / 16;   // 48
  const int NT = Dc / 16;          // 16
  int wave = wave_id();
  int lane = threadIdx.x & 31;
  if (wave >= 3 * MT * NT) return;
  int which = wave / (MT * NT);
  int t = wave % (MT * NT);
  int mt = t / NT, nt = t % NT;
  const float* W  = (which == 0) ? Wq : (which == 1) ? Wk : Wv;
  const float* bb = (which == 0) ? bq : (which == 1) ? bk : bv;
  float* out      = (which == 0) ? q  : (which == 1) ? k  : v;

  v8f acc = {};
  for (int k0 = 0; k0 < Dc; k0 += 32) {
    v16bf a = load_a_frag(h + (long)(mt * 16) * Dc + k0, Dc, lane);
    v16bf b = load_bT_frag(W + (long)(nt * 16) * Dc + k0, Dc, lane);
    acc = wmma_bf16(a, b, acc);
  }
  int n = lane & 15, mb = (lane & 16) ? 8 : 0;
  float bn = bb[nt * 16 + n];
  float* o = out + (long)(mt * 16 + mb) * Dc + nt * 16 + n;
#pragma unroll
  for (int r = 0; r < 8; ++r) o[(long)r * Dc] = acc[r] + bn;
}

// --------- Kernel 2: edge bias  [B*N*N,256] x We[8,256]^T -> logits ---------
// Streams rel_emb exactly once: coalesced 512B/instruction non-temporal loads,
// fp32->bf16 convert, staged in a per-wave 8KB LDS slab, A-fragments read back
// with ds_load_b128 (same-wave LDS ops are DScnt-ordered; no barrier needed).
__global__ void __launch_bounds__(128)
ebias_kernel(const float* __restrict__ rel,
             const float* __restrict__ We,
             const float* __restrict__ be,
             float* __restrict__ logits) {
  __shared__ __bf16 smem[WPB * 16 * Dc];           // 4 waves x 8KB
  const int MT = (Bc * Nc * Nc) / 16;              // 18432
  int wave = wave_id();
  int lane = threadIdx.x & 31;
  int wl   = (threadIdx.x >> 5) & (WPB - 1);
  if (wave >= MT) return;
  long row0 = (long)wave * 16;
  const float* A = rel + row0 * Dc;
  __bf16* sm = &smem[wl * 16 * Dc];

  // Stage 16x256 fp32 tile -> bf16 LDS, fully coalesced + non-temporal.
  const f32x4* Av = (const f32x4*)A;               // 1024 float4s
#pragma unroll 8
  for (int i = 0; i < 32; ++i) {
    int idx = i * 32 + lane;
    f32x4 d = __builtin_nontemporal_load(&Av[idx]);
    v4bf c;
    c[0] = (__bf16)d[0]; c[1] = (__bf16)d[1];
    c[2] = (__bf16)d[2]; c[3] = (__bf16)d[3];
    ((v4bf*)sm)[idx] = c;                          // ds_store_b64
  }

  int n  = lane & 15;
  int ko = (lane & 16) ? 16 : 0;
  int m  = lane & 15;
  int kb = (lane & 16) ? 16 : 0;

  v8f acc = {};
  for (int k0 = 0; k0 < Dc; k0 += 32) {
    // A fragment from LDS: two ds_load_b128 per K-step.
    v8bf lo = *(const v8bf*)(sm + m * Dc + k0 + kb);
    v8bf hi = *(const v8bf*)(sm + m * Dc + k0 + 8 + kb);
    v16bf a;
#pragma unroll
    for (int j = 0; j < 8; ++j) { a[j] = lo[j]; a[8 + j] = hi[j]; }
    v16bf b;
    if (n < Hc) {                                  // We is L2/L0-resident
      const float* r = We + (long)n * Dc + k0 + ko;
      f32x4 q0 = *(const f32x4*)(r + 0);
      f32x4 q1 = *(const f32x4*)(r + 4);
      f32x4 q2 = *(const f32x4*)(r + 8);
      f32x4 q3 = *(const f32x4*)(r + 12);
#pragma unroll
      for (int j = 0; j < 4; ++j) {
        b[j] = (__bf16)q0[j]; b[4 + j] = (__bf16)q1[j];
        b[8 + j] = (__bf16)q2[j]; b[12 + j] = (__bf16)q3[j];
      }
    } else {
#pragma unroll
      for (int e = 0; e < 16; ++e) b[e] = (__bf16)0.0f;
    }
    acc = wmma_bf16(a, b, acc);
  }

  int mb = (lane & 16) ? 8 : 0;
  if (n < Hc) {
    float bn = be[n];
#pragma unroll
    for (int r = 0; r < 8; ++r) {
      long row = row0 + mb + r;          // = (b*N + i)*N + j
      int  j  = (int)(row % Nc);
      long t2 = row / Nc;
      int  i  = (int)(t2 % Nc);
      int  bi = (int)(t2 / Nc);
      logits[((((long)bi * Hc + n) * Nc + i) * Nc) + j] = acc[r] + bn;
    }
  }
}

// --------- Kernel 3: logits = q.k^T * scale + bias, dual-masked -------------
__global__ void logits_kernel(const float* __restrict__ q,
                              const float* __restrict__ k,
                              const int* __restrict__ edge_mask,
                              const int* __restrict__ node_mask,
                              float* __restrict__ logits) {
  const int T = Nc / 16;               // 24
  int wave = wave_id();
  int lane = threadIdx.x & 31;
  if (wave >= Bc * Hc * T * T) return;
  int jt = wave % T; int t = wave / T;
  int it = t % T;    t /= T;
  int hh = t % Hc;   int bi = t / Hc;

  v16bf a = load_a_frag(q + (long)(bi * Nc + it * 16) * Dc + hh * HDc, Dc, lane);
  v16bf b = load_bT_frag(k + (long)(bi * Nc + jt * 16) * Dc + hh * HDc, Dc, lane);
  v8f acc = {};
  acc = wmma_bf16(a, b, acc);   // K = Hd = 32: one WMMA

  const float SC = 0.17677669529663687f;  // 32^-0.5
  int n = lane & 15, mb = (lane & 16) ? 8 : 0;
  int j = jt * 16 + n;
  bool keyok = node_mask[bi * Nc + j] != 0;
#pragma unroll
  for (int r = 0; r < 8; ++r) {
    int i = it * 16 + mb + r;
    long idx = (((long)bi * Hc + hh) * Nc + i) * Nc + j;
    bool valid = keyok && (edge_mask[((long)bi * Nc + i) * Nc + j] != 0);
    logits[idx] = valid ? (acc[r] * SC + logits[idx]) : -__builtin_inff();
  }
}

// --------- Kernel 4: row softmax (384 keys, one wave32 per row) -------------
__global__ void softmax_kernel(float* __restrict__ logits) {
  int wave = wave_id();
  int lane = threadIdx.x & 31;
  if (wave >= Bc * Hc * Nc) return;
  float* row = logits + (long)wave * Nc;
  float vals[12];
  float m = -__builtin_inff();
#pragma unroll
  for (int t = 0; t < 12; ++t) { vals[t] = row[lane + t * 32]; m = fmaxf(m, vals[t]); }
#pragma unroll
  for (int off = 16; off; off >>= 1) m = fmaxf(m, __shfl_xor(m, off, 32));
  float s = 0.f;
  if (m == -__builtin_inff()) {          // fully-masked row -> nan_to_num(0)
#pragma unroll
    for (int t = 0; t < 12; ++t) vals[t] = 0.f;
  } else {
#pragma unroll
    for (int t = 0; t < 12; ++t) { vals[t] = __expf(vals[t] - m); s += vals[t]; }
  }
#pragma unroll
  for (int off = 16; off; off >>= 1) s += __shfl_xor(s, off, 32);
  float inv = (s > 0.f) ? 1.f / s : 0.f;
#pragma unroll
  for (int t = 0; t < 12; ++t) row[lane + t * 32] = vals[t] * inv;
}

// --------- Kernel 5: out_head = attn @ v  (per b,h: 384x32, K=384) ----------
__global__ void av_kernel(const float* __restrict__ attn,
                          const float* __restrict__ v,
                          float* __restrict__ oh) {
  const int TI = Nc / 16;     // 24
  const int TN = HDc / 16;    // 2
  int wave = wave_id();
  int lane = threadIdx.x & 31;
  if (wave >= Bc * Hc * TI * TN) return;
  int nt = wave % TN; int t = wave / TN;
  int it = t % TI;    t /= TI;
  int hh = t % Hc;    int bi = t / Hc;

  const float* A  = attn + ((((long)bi * Hc + hh) * Nc + it * 16) * Nc);
  const float* Bm = v + (long)bi * Nc * Dc + hh * HDc + nt * 16;  // B[j][n], ld=Dc
  v8f acc = {};
  for (int k0 = 0; k0 < Nc; k0 += 32) {
    v16bf a = load_a_frag(A + k0, Nc, lane);
    v16bf b = load_b_rm_frag(Bm + (long)k0 * Dc, Dc, lane);
    acc = wmma_bf16(a, b, acc);
  }
  int n = lane & 15, mb = (lane & 16) ? 8 : 0;
  float* o = oh + (long)(bi * Nc + it * 16 + mb) * Dc + hh * HDc + nt * 16 + n;
#pragma unroll
  for (int r = 0; r < 8; ++r) o[(long)r * Dc] = acc[r];
}

// --------- Kernel 6: out = out_head @ Wo.T + bo -----------------------------
__global__ void oproj_kernel(const float* __restrict__ oh,
                             const float* __restrict__ Wo,
                             const float* __restrict__ bo,
                             float* __restrict__ out) {
  const int MT = (Bc * Nc) / 16, NT = Dc / 16;
  int wave = wave_id();
  int lane = threadIdx.x & 31;
  if (wave >= MT * NT) return;
  int mt = wave / NT, nt = wave % NT;
  v8f acc = {};
  for (int k0 = 0; k0 < Dc; k0 += 32) {
    v16bf a = load_a_frag(oh + (long)(mt * 16) * Dc + k0, Dc, lane);
    v16bf b = load_bT_frag(Wo + (long)(nt * 16) * Dc + k0, Dc, lane);
    acc = wmma_bf16(a, b, acc);
  }
  int n = lane & 15, mb = (lane & 16) ? 8 : 0;
  float bn = bo[nt * 16 + n];
  float* o = out + (long)(mt * 16 + mb) * Dc + nt * 16 + n;
#pragma unroll
  for (int r = 0; r < 8; ++r) o[(long)r * Dc] = acc[r] + bn;
}

// ---------------------------------------------------------------------------
extern "C" void kernel_launch(void* const* d_in, const int* in_sizes, int n_in,
                              void* d_out, int out_size, void* d_ws, size_t ws_size,
                              hipStream_t stream) {
  const float* h        = (const float*)d_in[0];
  const float* rel_emb  = (const float*)d_in[1];
  const int*   node_m   = (const int*)d_in[2];
  const int*   edge_m   = (const int*)d_in[3];
  const float* Wq = (const float*)d_in[4];  const float* bq = (const float*)d_in[5];
  const float* Wk = (const float*)d_in[6];  const float* bk = (const float*)d_in[7];
  const float* Wv = (const float*)d_in[8];  const float* bv = (const float*)d_in[9];
  const float* Wo = (const float*)d_in[10]; const float* bo = (const float*)d_in[11];
  const float* We = (const float*)d_in[12]; const float* be = (const float*)d_in[13];
  float* out = (float*)d_out;

  // Workspace layout (floats): q | k | v | logits | out_head  (~12.6 MB)
  float* ws = (float*)d_ws;
  const long QKV = (long)Bc * Nc * Dc;        // 196608
  float* q      = ws;
  float* k      = q + QKV;
  float* v      = k + QKV;
  float* logits = v + QKV;                    // B*H*N*N = 2359296
  float* oh     = logits + (long)Bc * Hc * Nc * Nc;
  (void)in_sizes; (void)n_in; (void)out_size; (void)ws_size;

  const int TPB = WPB * 32;  // 128 threads = 4 wave32
  auto blocks = [](int waves) { return (waves + WPB - 1) / WPB; };

  qkv_kernel<<<blocks(3 * 48 * 16), TPB, 0, stream>>>(h, Wq, bq, Wk, bk, Wv, bv, q, k, v);
  ebias_kernel<<<blocks((Bc * Nc * Nc) / 16), TPB, 0, stream>>>(rel_emb, We, be, logits);
  logits_kernel<<<blocks(Bc * Hc * 24 * 24), TPB, 0, stream>>>(q, k, edge_m, node_m, logits);
  softmax_kernel<<<blocks(Bc * Hc * Nc), TPB, 0, stream>>>(logits);
  av_kernel<<<blocks(Bc * Hc * 24 * 2), TPB, 0, stream>>>(logits, v, oh);
  oproj_kernel<<<blocks(48 * 16), TPB, 0, stream>>>(oh, Wo, bo, out);
}
